// Qwen3SparseMoeBlock_20830591386401
// MI455X (gfx1250) — matrix operand order repaired and checked by reading, not compiled, not run on hardware
//
#include <hip/hip_runtime.h>
#include <hip/hip_bf16.h>

// Problem constants (match reference)
#define T_TOK 8192   // B*S
#define H_DIM 2048
#define E_NUM 32
#define TOPK_ 8
#define IM    768
#define IS    4096

typedef __attribute__((ext_vector_type(8)))  float  v8f;
typedef __attribute__((ext_vector_type(16))) __bf16 v16bf;
typedef __attribute__((ext_vector_type(8)))  __bf16 v8bf;

// ---- CDNA5 async global->LDS copy (ASYNCcnt-tracked, no VGPR round trip) --
__device__ inline void async_copy_b64(void* lds_dst, const void* gsrc) {
  unsigned lds_off = (unsigned)(uintptr_t)lds_dst;           // low 32b = LDS offset
  unsigned long long ga = (unsigned long long)(uintptr_t)gsrc;
  asm volatile("global_load_async_to_lds_b64 %0, %1, off"
               :: "v"(lds_off), "v"(ga) : "memory");
}
__device__ inline void wait_async0() {
  asm volatile("s_wait_asynccnt 0x0" ::: "memory");
}

// ---- WMMA fragment loaders (wave32, per ISA 7.12.2) ----------------------
// A (16x32 bf16, row-major LDS tile, leading dim `ld`):
//   lanes 0-15: M=lane, K = k0+{0..7} then k0+{16..23}
//   lanes 16-31: M=lane-16, K = k0+{8..15} then k0+{24..31}
__device__ inline v16bf frag_a(const __bf16* sm, int ld, int mbase, int k0) {
  int lane = threadIdx.x & 31;
  int row  = mbase + (lane & 15);
  int kb   = k0 + ((lane < 16) ? 0 : 8);
  v16bf a;
#pragma unroll
  for (int i = 0; i < 8; ++i) {
    a[i]     = sm[row * ld + kb + i];
    a[i + 8] = sm[row * ld + kb + 16 + i];
  }
  return a;
}
// B (32x16 bf16), staged transposed in LDS as [n][32 k-values]:
//   lanes 0-15: N=lane, K={0..7} then {16..23} ; lanes 16-31: N=lane-16, K={8..15} then {24..31}
__device__ inline v16bf frag_b(const __bf16* smT, int nbase) {
  int lane = threadIdx.x & 31;
  int col  = nbase + (lane & 15);
  int kb   = (lane < 16) ? 0 : 8;
  v16bf b;
#pragma unroll
  for (int i = 0; i < 8; ++i) {
    b[i]     = smT[col * 32 + kb + i];
    b[i + 8] = smT[col * 32 + kb + 16 + i];
  }
  return b;
}

#define WMMA_BF16(A, B, C) \
  __builtin_amdgcn_wmma_f32_16x16x32_bf16(false, (A), false, (B), (short)0, (C), false, false)

// ---- fp32 -> bf16 activation conversion (vectorized) ---------------------
__global__ void k_cvt_bf16(const float* __restrict__ x, __bf16* __restrict__ xb) {
  size_t i = ((size_t)blockIdx.x * blockDim.x + threadIdx.x) * 8;
  float4 a = *(const float4*)(x + i);
  float4 b = *(const float4*)(x + i + 4);
  v8bf o;
  o[0] = (__bf16)a.x; o[1] = (__bf16)a.y; o[2] = (__bf16)a.z; o[3] = (__bf16)a.w;
  o[4] = (__bf16)b.x; o[5] = (__bf16)b.y; o[6] = (__bf16)b.z; o[7] = (__bf16)b.w;
  *(v8bf*)(xb + i) = o;
}

// ---- router: logits, softmax, top-8, renorm, per-expert token lists ------
__global__ void __launch_bounds__(256) k_router(
    const float* __restrict__ x, const float* __restrict__ gw,
    const float* __restrict__ sgw, float* __restrict__ logits_out,
    float* __restrict__ gate_s, int* __restrict__ cnt,
    int* __restrict__ tokL, float* __restrict__ wgtL) {
  const int wave = threadIdx.x >> 5;
  const int lane = threadIdx.x & 31;
  const int t    = blockIdx.x * 8 + wave;
  __shared__ float sprob[8][E_NUM];

  const float* xr = x + (size_t)t * H_DIM;
  // one lane per expert column
  float acc = 0.f;
  for (int k = 0; k < H_DIM; ++k) acc = fmaf(xr[k], gw[k * E_NUM + lane], acc);
  logits_out[t * E_NUM + lane] = acc;

  // softmax across the 32 lanes (fp32)
  float m = acc;
  for (int o = 16; o > 0; o >>= 1) m = fmaxf(m, __shfl_xor(m, o, 32));
  float p = __expf(acc - m);
  float s = p;
  for (int o = 16; o > 0; o >>= 1) s += __shfl_xor(s, o, 32);
  p /= s;
  sprob[wave][lane] = p;

  // shared-expert sigmoid gate: dot(x, sgw)
  float sd = 0.f;
  for (int k = lane; k < H_DIM; k += 32) sd = fmaf(xr[k], sgw[k], sd);
  for (int o = 16; o > 0; o >>= 1) sd += __shfl_xor(sd, o, 32);

  if (lane == 0) {
    gate_s[t] = 1.f / (1.f + __expf(-sd));
    int   idx[TOPK_]; float w[TOPK_]; float ws = 0.f;
    for (int j = 0; j < TOPK_; ++j) {
      float best = -1.f; int bi = 0;
      for (int e2 = 0; e2 < E_NUM; ++e2)
        if (sprob[wave][e2] > best) { best = sprob[wave][e2]; bi = e2; }
      idx[j] = bi; w[j] = best; ws += best; sprob[wave][bi] = -2.f;
    }
    for (int j = 0; j < TOPK_; ++j) {
      int e2 = idx[j];
      int pos = atomicAdd(&cnt[e2], 1);
      tokL[e2 * T_TOK + pos] = t;
      wgtL[e2 * T_TOK + pos] = w[j] / ws;   // norm_topk_prob
    }
  }
}

// ---- fused routed expert MLP: async gather -> SwiGLU (LDS-resident mid) ->
// ---- down-proj -> weighted atomic scatter --------------------------------
__global__ void __launch_bounds__(256) k_moe(
    const __bf16* __restrict__ xb, const float* __restrict__ Wg,
    const float* __restrict__ Wu, const float* __restrict__ Wd,
    const int* __restrict__ cnt, const int* __restrict__ tokL,
    const float* __restrict__ wgtL, float* __restrict__ out) {
  __shared__ int    tokS[32];
  __shared__ float  wS[32];
  __shared__ __bf16 smA[32 * 32];
  __shared__ __bf16 smBg[64 * 32];
  __shared__ __bf16 smBu[64 * 32];
  __shared__ __bf16 hmid[32 * IM];   // 48 KB: full SwiGLU mid for 32 rows

  const int e   = blockIdx.y;
  const int n_e = cnt[e];
  if ((int)blockIdx.x * 32 >= n_e) return;

  const int tid = threadIdx.x;
  if (tid < 32) {
    int g = blockIdx.x * 32 + tid;
    if (g < n_e) { tokS[tid] = tokL[e * T_TOK + g]; wS[tid] = wgtL[e * T_TOK + g]; }
    else         { tokS[tid] = -1;                  wS[tid] = 0.f; }
  }
  __syncthreads();

  const int wv   = tid >> 5;
  const int m2   = (wv >> 2) * 16;  // 0 / 16
  const int n4   = (wv & 3) * 16;   // 0..48
  const int lane = tid & 31;
  const int r  = tid >> 3;          // staging row 0..31
  const int k4 = (tid & 7) * 4;     // staging col (4 bf16 = 8B)
  const int myTok = tokS[r];
  const float* wgE = Wg + (size_t)e * H_DIM * IM;
  const float* wuE = Wu + (size_t)e * H_DIM * IM;
  const float* wdE = Wd + (size_t)e * IM * H_DIM;

  // --- GEMM1: [32,H] x [H,64]-tiles of Wg/Wu, SwiGLU into LDS mid ---
  for (int nt = 0; nt < IM; nt += 64) {
    v8f cg = {}; v8f cu = {};
    for (int k0 = 0; k0 < H_DIM; k0 += 32) {
      // stage gathered A rows via CDNA5 async global->LDS copy
      if (myTok >= 0) {
        async_copy_b64(&smA[r * 32 + k4],
                       xb + (size_t)myTok * H_DIM + k0 + k4);
      } else {
#pragma unroll
        for (int j = 0; j < 4; ++j) smA[r * 32 + k4 + j] = (__bf16)0.f;
      }
#pragma unroll
      for (int i = 0; i < 8; ++i) {  // stage B transposed, cvt fp32->bf16
        int lin = tid + i * 256;
        int n = lin & 63, kk = lin >> 6;
        size_t gi = (size_t)(k0 + kk) * IM + nt + n;
        smBg[n * 32 + kk] = (__bf16)wgE[gi];
        smBu[n * 32 + kk] = (__bf16)wuE[gi];
        if (k0 + 32 < H_DIM) {       // prefetch next K-step weight rows
          __builtin_prefetch(&wgE[gi + 32 * IM], 0, 0);
          __builtin_prefetch(&wuE[gi + 32 * IM], 0, 0);
        }
      }
      wait_async0();
      __syncthreads();
      v16bf a  = frag_a(smA, 32, m2, 0);
      v16bf bg = frag_b(smBg, n4);
      v16bf bu = frag_b(smBu, n4);
      cg = WMMA_BF16(a, bg, cg);
      cu = WMMA_BF16(a, bu, cu);
      __syncthreads();
    }
#pragma unroll
    for (int j = 0; j < 8; ++j) {    // silu(g)*u -> LDS mid (bf16)
      int mr = m2 + j + ((lane >= 16) ? 8 : 0);
      int c  = nt + n4 + (lane & 15);
      float g = cg[j], u = cu[j];
      hmid[mr * IM + c] = (__bf16)((g / (1.f + __expf(-g))) * u);
    }
    __syncthreads();
  }

  // --- GEMM2: out += w_row * (hmid @ Wd), A straight from LDS ---
  for (int nt = 0; nt < H_DIM; nt += 64) {
    v8f co = {};
    for (int k0 = 0; k0 < IM; k0 += 32) {
#pragma unroll
      for (int i = 0; i < 8; ++i) {
        int lin = tid + i * 256;
        int n = lin & 63, kk = lin >> 6;
        smBg[n * 32 + kk] = (__bf16)wdE[(size_t)(k0 + kk) * H_DIM + nt + n];
      }
      __syncthreads();
      v16bf a = frag_a(hmid, IM, m2, k0);
      v16bf b = frag_b(smBg, n4);
      co = WMMA_BF16(a, b, co);
      __syncthreads();
    }
#pragma unroll
    for (int j = 0; j < 8; ++j) {
      int mr = m2 + j + ((lane >= 16) ? 8 : 0);
      int tkn = tokS[mr];
      if (tkn >= 0)
        atomicAdd(&out[(size_t)tkn * H_DIM + nt + n4 + (lane & 15)], co[j] * wS[mr]);
    }
  }
}

// ---- shared expert, pass 1: hs = bf16(silu(x@Sg) * (x@Su)) ---------------
__global__ void __launch_bounds__(256) k_shared1(
    const __bf16* __restrict__ xb, const float* __restrict__ Sg,
    const float* __restrict__ Su, __bf16* __restrict__ hs) {
  __shared__ __bf16 smA[32 * 32], smBg[64 * 32], smBu[64 * 32];
  const int tid = threadIdx.x, wv = tid >> 5, lane = tid & 31;
  const int m2 = (wv >> 2) * 16, n4 = (wv & 3) * 16;
  const int t0 = blockIdx.x * 32, nt = blockIdx.y * 64;
  const int r = tid >> 3, k4 = (tid & 7) * 4;
  v8f cg = {}, cu = {};
  for (int k0 = 0; k0 < H_DIM; k0 += 32) {
    async_copy_b64(&smA[r * 32 + k4],
                   xb + (size_t)(t0 + r) * H_DIM + k0 + k4);
#pragma unroll
    for (int i = 0; i < 8; ++i) {
      int lin = tid + i * 256;
      int n = lin & 63, kk = lin >> 6;
      size_t gi = (size_t)(k0 + kk) * IS + nt + n;
      smBg[n * 32 + kk] = (__bf16)Sg[gi];
      smBu[n * 32 + kk] = (__bf16)Su[gi];
    }
    wait_async0();
    __syncthreads();
    v16bf a = frag_a(smA, 32, m2, 0);
    cg = WMMA_BF16(a, frag_b(smBg, n4), cg);
    cu = WMMA_BF16(a, frag_b(smBu, n4), cu);
    __syncthreads();
  }
#pragma unroll
  for (int j = 0; j < 8; ++j) {
    int mr = m2 + j + ((lane >= 16) ? 8 : 0);
    int c  = nt + n4 + (lane & 15);
    float g = cg[j], u = cu[j];
    hs[(size_t)(t0 + mr) * IS + c] = (__bf16)((g / (1.f + __expf(-g))) * u);
  }
}

// ---- shared expert, pass 2: out = sigmoid_gate * (hs @ Sd)  (initializes out)
__global__ void __launch_bounds__(256) k_shared2(
    const __bf16* __restrict__ hs, const float* __restrict__ Sd,
    const float* __restrict__ gate_s, float* __restrict__ out) {
  __shared__ __bf16 smA[32 * 32], smB[64 * 32];
  const int tid = threadIdx.x, wv = tid >> 5, lane = tid & 31;
  const int m2 = (wv >> 2) * 16, n4 = (wv & 3) * 16;
  const int t0 = blockIdx.x * 32, nt = blockIdx.y * 64;
  const int r = tid >> 3, k4 = (tid & 7) * 4;
  v8f co = {};
  for (int k0 = 0; k0 < IS; k0 += 32) {
    async_copy_b64(&smA[r * 32 + k4],
                   hs + (size_t)(t0 + r) * IS + k0 + k4);
#pragma unroll
    for (int i = 0; i < 8; ++i) {
      int lin = tid + i * 256;
      int n = lin & 63, kk = lin >> 6;
      smB[n * 32 + kk] = (__bf16)Sd[(size_t)(k0 + kk) * H_DIM + nt + n];
    }
    wait_async0();
    __syncthreads();
    co = WMMA_BF16(frag_a(smA, 32, m2, 0), frag_b(smB, n4), co);
    __syncthreads();
  }
#pragma unroll
  for (int j = 0; j < 8; ++j) {
    int mr = m2 + j + ((lane >= 16) ? 8 : 0);
    int t  = t0 + mr;
    out[(size_t)t * H_DIM + nt + n4 + (lane & 15)] = gate_s[t] * co[j];
  }
}

extern "C" void kernel_launch(void* const* d_in, const int* in_sizes, int n_in,
                              void* d_out, int out_size, void* d_ws, size_t ws_size,
                              hipStream_t stream) {
  (void)in_sizes; (void)n_in; (void)out_size; (void)ws_size;
  const float* x   = (const float*)d_in[0];
  const float* gw  = (const float*)d_in[1];
  const float* Wg  = (const float*)d_in[2];
  const float* Wu  = (const float*)d_in[3];
  const float* Wd  = (const float*)d_in[4];
  const float* Sg  = (const float*)d_in[5];
  const float* Su  = (const float*)d_in[6];
  const float* Sd  = (const float*)d_in[7];
  const float* sgw = (const float*)d_in[8];

  float* out_final  = (float*)d_out;                       // [T,H]
  float* out_logits = out_final + (size_t)T_TOK * H_DIM;   // [T,E]

  char* ws = (char*)d_ws;
  __bf16* xb     = (__bf16*)ws;  ws += (size_t)T_TOK * H_DIM * sizeof(__bf16);
  __bf16* hs     = (__bf16*)ws;  ws += (size_t)T_TOK * IS   * sizeof(__bf16);
  float*  gate_s = (float*)ws;   ws += (size_t)T_TOK * sizeof(float);
  int*    cnt    = (int*)ws;     ws += 256;
  int*    tokL   = (int*)ws;     ws += (size_t)E_NUM * T_TOK * sizeof(int);
  float*  wgtL   = (float*)ws;

  hipMemsetAsync(cnt, 0, E_NUM * sizeof(int), stream);
  k_cvt_bf16<<<(T_TOK * H_DIM) / (256 * 8), 256, 0, stream>>>(x, xb);
  k_router<<<T_TOK / 8, 256, 0, stream>>>(x, gw, sgw, out_logits, gate_s, cnt, tokL, wgtL);
  k_shared1<<<dim3(T_TOK / 32, IS / 64), 256, 0, stream>>>(xb, Sg, Su, hs);
  k_shared2<<<dim3(T_TOK / 32, H_DIM / 64), 256, 0, stream>>>(hs, Sd, gate_s, out_final);
  k_moe<<<dim3(T_TOK / 32, E_NUM), 256, 0, stream>>>(xb, Wg, Wu, Wd, cnt, tokL, wgtL, out_final);
}
